// EncoderBlock_87222195847962
// MI455X (gfx1250) — matrix operand order repaired and checked by reading, not compiled, not run on hardware
//
#include <hip/hip_runtime.h>
#include <hip/hip_bf16.h>

typedef __attribute__((ext_vector_type(16))) _Float16 v16h;
typedef __attribute__((ext_vector_type(8)))  _Float16 v8h;
typedef __attribute__((ext_vector_type(2)))  _Float16 v2h;
typedef __attribute__((ext_vector_type(8)))  float    v8f;

#define BB 16
#define DD 256
#define LL 1024
#define NH 8
#define KW 7
#define NPB (DD*LL)           /* 262144 per-batch elements */
#define NTOT (BB*DD*LL)       /* 4194304 */

__device__ inline v8f wmma16(v16h a, v16h b, v8f c) {
  return __builtin_amdgcn_wmma_f32_16x16x32_f16(false, a, false, b, (short)0, c, false, false);
}

__device__ inline v16h cat16(v8h lo, v8h hi) {
  return __builtin_shufflevector(lo, hi, 0,1,2,3,4,5,6,7,8,9,10,11,12,13,14,15);
}

// A fragment from row-major [row][k] LDS tile (ld in halfs; row stride 16B-aligned).
// lane<16: row0+lane, K={0..7}U{16..23}; lane>=16: K={8..15}U{24..31}  (ISA 7.12.2)
__device__ inline v16h frag_a(const _Float16* s, int row0, int ld) {
  int lane = threadIdx.x & 31;
  const _Float16* p = s + (row0 + (lane & 15)) * ld + ((lane >> 4) << 3);
  v8h lo = *(const v8h*)p;
  v8h hi = *(const v8h*)(p + 16);
  return cat16(lo, hi);
}
// B fragment from TRANSPOSED [n][k] LDS tile: lanes 0-15 K={0..7}U{16..23} of col n,
// lanes 16-31 K={8..15}U{24..31} -> two b128 loads.
__device__ inline v16h frag_bt(const _Float16* s, int col0, int ld) {
  int lane = threadIdx.x & 31;
  const _Float16* p = s + (col0 + (lane & 15)) * ld + ((lane >> 4) << 3);
  v8h lo = *(const v8h*)p;
  v8h hi = *(const v8h*)(p + 16);
  return cat16(lo, hi);
}

__device__ inline float red16_max(float v) {
#pragma unroll
  for (int m = 1; m < 16; m <<= 1) v = fmaxf(v, __shfl_xor(v, m, 32));
  return v;
}
__device__ inline float red16_sum(float v) {
#pragma unroll
  for (int m = 1; m < 16; m <<= 1) v += __shfl_xor(v, m, 32);
  return v;
}
__device__ inline float wave_sum(float v) {
#pragma unroll
  for (int m = 1; m < 32; m <<= 1) v += __shfl_xor(v, m, 32);
  return v;
}

// ---------------------------------------------------------------- utilities
__global__ void zero_stats_kernel(float* stats) {
  if (threadIdx.x < 32) stats[threadIdx.x] = 0.f;
}

__global__ void add_pos_kernel(const float* __restrict__ x, float* __restrict__ res,
                               float* __restrict__ stats) {
  __shared__ float red1[8], red2[8];
  int i = blockIdx.x * 256 + threadIdx.x;
  int l = i & (LL - 1);
  int d = (i >> 10) & (DD - 1);
  int batch = i >> 18;
  float pe;
  if ((d & 1) == 0)
    pe = sinf((float)l * powf(10000.f, -(float)d / 256.f));
  else
    pe = cosf((float)l * powf(10000.f, (1.f - (float)d) / 256.f));
  float v = x[i] + pe;
  res[i] = v;
  float s1 = wave_sum(v), s2 = wave_sum(v * v);
  int wave = threadIdx.x >> 5, lane = threadIdx.x & 31;
  if (lane == 0) { red1[wave] = s1; red2[wave] = s2; }
  __syncthreads();
  if (threadIdx.x == 0) {
    float a = 0.f, b2 = 0.f;
    for (int w = 0; w < 8; ++w) { a += red1[w]; b2 += red2[w]; }
    atomicAdd(&stats[batch * 2 + 0], a);
    atomicAdd(&stats[batch * 2 + 1], b2);
  }
}

// 8 elements per thread, b128 store
__global__ void ln_kernel(const float* __restrict__ res, const float* __restrict__ stats,
                          const float* __restrict__ g, const float* __restrict__ bta,
                          _Float16* __restrict__ outh) {
  int u = blockIdx.x * 256 + threadIdx.x;   // over NTOT/8
  size_t i0 = (size_t)u << 3;
  int batch = (int)(i0 >> 18);
  int dl = (int)(i0 & (NPB - 1));
  float mean = stats[batch * 2] * (1.f / (float)NPB);
  float var  = stats[batch * 2 + 1] * (1.f / (float)NPB) - mean * mean;
  float rs = rsqrtf(var + 1e-5f);
  v8h o;
#pragma unroll
  for (int j = 0; j < 8; ++j)
    o[j] = (_Float16)((res[i0 + j] - mean) * rs * g[dl + j] + bta[dl + j]);
  *(v8h*)&outh[i0] = o;
}

// 8 outputs per thread along L
__global__ void dwconv_kernel(const _Float16* __restrict__ inh, const float* __restrict__ w,
                              const float* __restrict__ bias, _Float16* __restrict__ outh) {
  int u = blockIdx.x * 256 + threadIdx.x;   // over NTOT/8
  int l8 = (u & 127) << 3;
  int d = (u >> 7) & 255;
  int bidx = u >> 15;
  size_t base = ((size_t)bidx * DD + d) * LL;
  float wv[KW];
#pragma unroll
  for (int t = 0; t < KW; ++t) wv[t] = w[d * KW + t];
  float xr[14];
#pragma unroll
  for (int t = 0; t < 14; ++t) {
    int l = l8 + t - 3;
    xr[t] = (l >= 0 && l < LL) ? (float)inh[base + l] : 0.f;
  }
  float bb = bias[d];
  v8h o;
#pragma unroll
  for (int j = 0; j < 8; ++j) {
    float acc = bb;
#pragma unroll
    for (int t = 0; t < KW; ++t) acc += wv[t] * xr[j + t];
    o[j] = (_Float16)acc;
  }
  *(v8h*)&outh[base + l8] = o;
}

__global__ void convert_w_kernel(const float* __restrict__ src, _Float16* __restrict__ dst) {
  int i = blockIdx.x * 256 + threadIdx.x;
  dst[i] = (_Float16)src[i];
}
// Wq/Wk/Wv: (Nh, D, Dk) -> A[(h*32+k), d]
__global__ void convert_qkv_w_kernel(const float* __restrict__ src, _Float16* __restrict__ dst) {
  int i = blockIdx.x * 256 + threadIdx.x;   // over 65536
  int kk = i & 31;
  int d  = (i >> 5) & 255;
  int h  = i >> 13;
  dst[(h * 32 + kk) * DD + d] = (_Float16)src[i];
}
// Wo: (j, e) -> A[e, j]
__global__ void transpose_w_kernel(const float* __restrict__ src, _Float16* __restrict__ dst) {
  int i = blockIdx.x * 256 + threadIdx.x;   // dst linear e*256+j
  int e = i >> 8, j = i & 255;
  dst[i] = (_Float16)src[j * DD + e];
}

// ------------------------------------------------ GEMM core (shared by both kernels)
// C(64x64 per block) = A(256x256) x B(b)(256x1024); double-buffered LDS, B transposed.
#define GEMM_MAINLOOP(ACC)                                                          \
  const int ar0 = tid >> 2;                                                         \
  const int ac0 = (tid & 3) << 3;                                                   \
  const int bkp = tid >> 3;                                                         \
  const int bn8 = (tid & 7) << 3;                                                   \
  v8h aR0, aR1, b0R, b1R;                                                           \
  {                                                                                 \
    aR0 = *(const v8h*)&A[(size_t)(M0 + ar0) * DD + ac0];                           \
    aR1 = *(const v8h*)&A[(size_t)(M0 + 32 + ar0) * DD + ac0];                      \
    b0R = *(const v8h*)&Bb[(size_t)(2 * bkp) * LL + N0 + bn8];                      \
    b1R = *(const v8h*)&Bb[(size_t)(2 * bkp + 1) * LL + N0 + bn8];                  \
    *(v8h*)&As[0][ar0][ac0] = aR0;                                                  \
    *(v8h*)&As[0][32 + ar0][ac0] = aR1;                                             \
    _Pragma("unroll")                                                               \
    for (int j = 0; j < 8; ++j) {                                                   \
      v2h t; t[0] = b0R[j]; t[1] = b1R[j];                                          \
      *(v2h*)&Bs[0][bn8 + j][2 * bkp] = t;                                          \
    }                                                                               \
  }                                                                                 \
  __syncthreads();                                                                  \
  int cur = 0;                                                                      \
  for (int ks = 0; ks < 8; ++ks) {                                                  \
    int k0n = (ks + 1) * 32;                                                        \
    if (ks < 7) {                                                                   \
      aR0 = *(const v8h*)&A[(size_t)(M0 + ar0) * DD + k0n + ac0];                   \
      aR1 = *(const v8h*)&A[(size_t)(M0 + 32 + ar0) * DD + k0n + ac0];              \
      b0R = *(const v8h*)&Bb[(size_t)(k0n + 2 * bkp) * LL + N0 + bn8];              \
      b1R = *(const v8h*)&Bb[(size_t)(k0n + 2 * bkp + 1) * LL + N0 + bn8];          \
    }                                                                               \
    v16h a0 = frag_a(&As[cur][0][0], wm, 40);                                       \
    v16h a1 = frag_a(&As[cur][0][0], wm + 16, 40);                                  \
    v16h b0 = frag_bt(&Bs[cur][0][0], wn, 40);                                      \
    v16h b1 = frag_bt(&Bs[cur][0][0], wn + 16, 40);                                 \
    ACC[0][0] = wmma16(a0, b0, ACC[0][0]);                                          \
    ACC[0][1] = wmma16(a0, b1, ACC[0][1]);                                          \
    ACC[1][0] = wmma16(a1, b0, ACC[1][0]);                                          \
    ACC[1][1] = wmma16(a1, b1, ACC[1][1]);                                          \
    if (ks < 7) {                                                                   \
      int nb = cur ^ 1;                                                             \
      *(v8h*)&As[nb][ar0][ac0] = aR0;                                               \
      *(v8h*)&As[nb][32 + ar0][ac0] = aR1;                                          \
      _Pragma("unroll")                                                             \
      for (int j = 0; j < 8; ++j) {                                                 \
        v2h t; t[0] = b0R[j]; t[1] = b1R[j];                                        \
        *(v2h*)&Bs[nb][bn8 + j][2 * bkp] = t;                                       \
      }                                                                             \
    }                                                                               \
    __syncthreads();                                                                \
    cur ^= 1;                                                                       \
  }

// epilogue: (+bias)(+relu) + resid -> out(f32), optional LN-stat atomics
__global__ void __launch_bounds__(128)
gemm_resid_kernel(const _Float16* __restrict__ A, const _Float16* __restrict__ Bm,
                  const float* __restrict__ bias, const float* resid, float* out,
                  float* stats, int relu) {
  __shared__ __align__(16) _Float16 As[2][64][40];
  __shared__ __align__(16) _Float16 Bs[2][64][40];   // transposed [n][k]
  __shared__ float red1[4], red2[4];
  const int tid = threadIdx.x;
  const int batch = blockIdx.z;
  const int M0 = blockIdx.y * 64;
  const int N0 = blockIdx.x * 64;
  const _Float16* Bb = Bm + (size_t)batch * NPB;
  const int wave = tid >> 5;
  const int wm = (wave >> 1) * 32;
  const int wn = (wave & 1) * 32;
  v8f acc[2][2] = {};
  GEMM_MAINLOOP(acc)

  const int lane = tid & 31;
  const int hs = lane >> 4, nc = lane & 15;
  float bv[2][8];
#pragma unroll
  for (int mi = 0; mi < 2; ++mi)
#pragma unroll
    for (int r = 0; r < 8; ++r) bv[mi][r] = 0.f;
  if (bias) {
#pragma unroll
    for (int mi = 0; mi < 2; ++mi)
#pragma unroll
      for (int r = 0; r < 8; ++r) bv[mi][r] = bias[M0 + wm + mi * 16 + hs * 8 + r];
  }
  float s1 = 0.f, s2 = 0.f;
#pragma unroll
  for (int mi = 0; mi < 2; ++mi)
#pragma unroll
    for (int ni = 0; ni < 2; ++ni) {
      size_t base = ((size_t)batch * DD + (M0 + wm + mi * 16 + hs * 8)) * LL
                  + (N0 + wn + ni * 16 + nc);
      const float* rp = resid + base;
      float* op = out + base;
#pragma unroll
      for (int r = 0; r < 8; ++r) {
        float v = acc[mi][ni][r] + bv[mi][r];
        if (relu) v = fmaxf(v, 0.f);
        v += rp[(size_t)r * LL];
        op[(size_t)r * LL] = v;
        s1 += v; s2 += v * v;
      }
    }
  if (stats) {
    s1 = wave_sum(s1); s2 = wave_sum(s2);
    if (lane == 0) { red1[wave] = s1; red2[wave] = s2; }
    __syncthreads();
    if (tid == 0) {
      float a = 0.f, b2 = 0.f;
      for (int w = 0; w < 4; ++w) { a += red1[w]; b2 += red2[w]; }
      atomicAdd(&stats[batch * 2 + 0], a);
      atomicAdd(&stats[batch * 2 + 1], b2);
    }
  }
}

// half output (QKV projections): out = half(acc * scale)
__global__ void __launch_bounds__(128)
gemm_half_kernel(const _Float16* __restrict__ A, const _Float16* __restrict__ Bm,
                 _Float16* __restrict__ outh, float scale) {
  __shared__ __align__(16) _Float16 As[2][64][40];
  __shared__ __align__(16) _Float16 Bs[2][64][40];
  const int tid = threadIdx.x;
  const int batch = blockIdx.z;
  const int M0 = blockIdx.y * 64;
  const int N0 = blockIdx.x * 64;
  const _Float16* Bb = Bm + (size_t)batch * NPB;
  const int wave = tid >> 5;
  const int wm = (wave >> 1) * 32;
  const int wn = (wave & 1) * 32;
  v8f acc[2][2] = {};
  GEMM_MAINLOOP(acc)

  const int lane = tid & 31;
  const int hs = lane >> 4, nc = lane & 15;
#pragma unroll
  for (int mi = 0; mi < 2; ++mi)
#pragma unroll
    for (int ni = 0; ni < 2; ++ni) {
      size_t base = ((size_t)batch * DD + (M0 + wm + mi * 16 + hs * 8)) * LL
                  + (N0 + wn + ni * 16 + nc);
      _Float16* hp = outh + base;
#pragma unroll
      for (int r = 0; r < 8; ++r)
        hp[(size_t)r * LL] = (_Float16)(acc[mi][ni][r] * scale);
    }
}

// ------------------------------------------------ flash attention
// q/k/v: (B, Nh*32, L) f16, 1/sqrt(dk) folded into q. heads out: (B, Nh*32, L) f16.
// grid (L/128, Nh, B), block 128; 32 queries/wave, key tile 32, double-buffered K/V.
__global__ void __launch_bounds__(128)
attn_kernel(const _Float16* __restrict__ qh, const _Float16* __restrict__ kh,
            const _Float16* __restrict__ vh, const int* __restrict__ mask,
            _Float16* __restrict__ headsh) {
  __shared__ __align__(16) _Float16 Qs[128][40];       // [q][dk] (A layout); reused for O
  __shared__ __align__(16) _Float16 Kt[2][32][40];     // [key][dk]  (transposed-B)
  __shared__ __align__(16) _Float16 Vt[2][32][40];     // [dv][key]  (transposed-B)
  __shared__ __align__(16) _Float16 Ps[4][32][40];     // per-wave P (A layout)
  __shared__ float kmaskS[2][32];
  __shared__ float qmask[128];

  const int b = blockIdx.z, h = blockIdx.y;
  const int q0 = blockIdx.x * 128;
  const int tid = threadIdx.x, wave = tid >> 5, lane = tid & 31;
  const int hs = lane >> 4, nc = lane & 15;
  const size_t hb = (size_t)b * DD + h * 32;
  const _Float16* qb  = qh + hb * LL;
  const _Float16* kbp = kh + hb * LL;
  const _Float16* vbp = vh + hb * LL;

  // stage Q (pack-transpose to [q][dk]): 16 dk-pairs x 16 q-chunks
#pragma unroll
  for (int it = 0; it < 2; ++it) {
    int u = it * 128 + tid;
    int cp = u >> 4, q8 = (u & 15) << 3;
    v8h r0 = *(const v8h*)&qb[(size_t)(2 * cp) * LL + q0 + q8];
    v8h r1 = *(const v8h*)&qb[(size_t)(2 * cp + 1) * LL + q0 + q8];
#pragma unroll
    for (int j = 0; j < 8; ++j) {
      v2h t; t[0] = r0[j]; t[1] = r1[j];
      *(v2h*)&Qs[q8 + j][2 * cp] = t;
    }
  }
  qmask[tid] = (float)mask[b * LL + q0 + tid];

  // K/V register staging (threads 0-63: K pack-transpose, 64-127: V direct copy)
  v8h k0R, k1R, v0R, v1R; float kmR = 0.f;
  const int kcp = tid >> 2, kkc = (tid & 3) << 3;          // K: dk-pair, key-chunk
  const int vp = tid - 64, vn = (tid - 64) >> 2, vc8 = ((tid - 64) & 3) << 3;
  auto kv_load = [&](int kb0) {
    if (tid < 64) {
      k0R = *(const v8h*)&kbp[(size_t)(2 * kcp) * LL + kb0 + kkc];
      k1R = *(const v8h*)&kbp[(size_t)(2 * kcp + 1) * LL + kb0 + kkc];
    } else {
      v0R = *(const v8h*)&vbp[(size_t)vn * LL + kb0 + vc8];
      v1R = *(const v8h*)&vbp[(size_t)(vn + 16) * LL + kb0 + vc8];
    }
    if (tid < 32) kmR = (float)mask[b * LL + kb0 + tid];
  };
  auto kv_store = [&](int buf) {
    if (tid < 64) {
#pragma unroll
      for (int j = 0; j < 8; ++j) {
        v2h t; t[0] = k0R[j]; t[1] = k1R[j];
        *(v2h*)&Kt[buf][kkc + j][2 * kcp] = t;
      }
    } else {
      *(v8h*)&Vt[buf][vn][vc8] = v0R;
      *(v8h*)&Vt[buf][vn + 16][vc8] = v1R;
    }
    if (tid < 32) kmaskS[buf][tid] = kmR;
  };

  kv_load(0);
  kv_store(0);
  __syncthreads();

  const int qrow0 = wave * 32;
  v16h aq0 = frag_a(&Qs[0][0], qrow0, 40);
  v16h aq1 = frag_a(&Qs[0][0], qrow0 + 16, 40);

  float mstate[2][8], lstate[2][8];
#pragma unroll
  for (int mi = 0; mi < 2; ++mi)
#pragma unroll
    for (int r = 0; r < 8; ++r) { mstate[mi][r] = -1e38f; lstate[mi][r] = 0.f; }
  v8f o[2][2] = {};

  int cur = 0;
  for (int it = 0; it < 32; ++it) {
    if (it < 31) kv_load((it + 1) * 32);

    v16h bk0 = frag_bt(&Kt[cur][0][0], 0, 40);
    v16h bk1 = frag_bt(&Kt[cur][0][0], 16, 40);
    v8f z = {};
    v8f s[2][2];
    s[0][0] = wmma16(aq0, bk0, z);
    s[0][1] = wmma16(aq0, bk1, z);
    s[1][0] = wmma16(aq1, bk0, z);
    s[1][1] = wmma16(aq1, bk1, z);

    // key mask
    float km0 = kmaskS[cur][nc], km1 = kmaskS[cur][16 + nc];
#pragma unroll
    for (int mi = 0; mi < 2; ++mi)
#pragma unroll
      for (int r = 0; r < 8; ++r) {
        s[mi][0][r] = (km0 != 0.f) ? s[mi][0][r] : -1e30f;
        s[mi][1][r] = (km1 != 0.f) ? s[mi][1][r] : -1e30f;
      }

    // online softmax per row (row = mi*16 + hs*8 + r)
#pragma unroll
    for (int mi = 0; mi < 2; ++mi)
#pragma unroll
      for (int r = 0; r < 8; ++r) {
        float rm = fmaxf(red16_max(s[mi][0][r]), red16_max(s[mi][1][r]));
        float mold = mstate[mi][r];
        float mnew = fmaxf(mold, rm);
        float cf = __expf(mold - mnew);
        mstate[mi][r] = mnew;
        float p0 = __expf(s[mi][0][r] - mnew);
        float p1 = __expf(s[mi][1][r] - mnew);
        s[mi][0][r] = p0;
        s[mi][1][r] = p1;
        lstate[mi][r] = lstate[mi][r] * cf + red16_sum(p0) + red16_sum(p1);
        o[mi][0][r] *= cf;
        o[mi][1][r] *= cf;
      }

    // P -> per-wave LDS to re-enter A layout (intra-wave: ds-counter wait only)
#pragma unroll
    for (int mi = 0; mi < 2; ++mi)
#pragma unroll
      for (int ni = 0; ni < 2; ++ni)
#pragma unroll
        for (int r = 0; r < 8; ++r)
          Ps[wave][mi * 16 + hs * 8 + r][ni * 16 + nc] = (_Float16)s[mi][ni][r];
    asm volatile("s_wait_dscnt 0x0" ::: "memory");

    v16h ap0 = frag_a(&Ps[wave][0][0], 0, 40);
    v16h ap1 = frag_a(&Ps[wave][0][0], 16, 40);
    v16h bv0 = frag_bt(&Vt[cur][0][0], 0, 40);
    v16h bv1 = frag_bt(&Vt[cur][0][0], 16, 40);
    o[0][0] = wmma16(ap0, bv0, o[0][0]);
    o[0][1] = wmma16(ap0, bv1, o[0][1]);
    o[1][0] = wmma16(ap1, bv0, o[1][0]);
    o[1][1] = wmma16(ap1, bv1, o[1][1]);

    if (it < 31) kv_store(cur ^ 1);
    __syncthreads();
    cur ^= 1;
  }

  // normalize + query mask, transpose through LDS (reuse Qs as [q][dv])
#pragma unroll
  for (int mi = 0; mi < 2; ++mi)
#pragma unroll
    for (int r = 0; r < 8; ++r) {
      int rowl = qrow0 + mi * 16 + hs * 8 + r;
      float qm = qmask[rowl];
      float inv = (qm != 0.f) ? 1.f / lstate[mi][r] : 0.f;
      Qs[rowl][nc]      = (_Float16)(o[mi][0][r] * inv);
      Qs[rowl][16 + nc] = (_Float16)(o[mi][1][r] * inv);
    }
  __syncthreads();
  // coalesced b128 stores: heads (b, h*32+dv, l)
#pragma unroll
  for (int it = 0; it < 4; ++it) {
    int u = it * 128 + tid;
    int n = u >> 4, q8 = (u & 15) << 3;
    v8h t;
#pragma unroll
    for (int j = 0; j < 8; ++j) t[j] = Qs[q8 + j][n];
    *(v8h*)&headsh[(hb + n) * (size_t)LL + q0 + q8] = t;
  }
}

// ---------------------------------------------------------------- launch
extern "C" void kernel_launch(void* const* d_in, const int* in_sizes, int n_in,
                              void* d_out, int out_size, void* d_ws, size_t ws_size,
                              hipStream_t stream) {
  const float* x    = (const float*)d_in[0];
  const int*   mask = (const int*)d_in[1];
  const float* dw_w = (const float*)d_in[2];
  const float* dw_b = (const float*)d_in[3];
  const float* pw_w = (const float*)d_in[4];
  const float* pw_b = (const float*)d_in[5];
  const float* ln_g = (const float*)d_in[6];
  const float* ln_b = (const float*)d_in[7];
  const float* Wq   = (const float*)d_in[8];
  const float* Wk   = (const float*)d_in[9];
  const float* Wv   = (const float*)d_in[10];
  const float* Wo   = (const float*)d_in[11];
  const float* fc_w = (const float*)d_in[12];
  const float* fc_b = (const float*)d_in[13];
  float* out = (float*)d_out;

  char* p = (char*)d_ws;
  auto alloc = [&](size_t bytes) -> void* {
    void* r = (void*)p;
    p += (bytes + 255) & ~(size_t)255;
    return r;
  };
  float*    res   = (float*)alloc((size_t)NTOT * 4);
  _Float16* curh  = (_Float16*)alloc((size_t)NTOT * 2);
  _Float16* dwh   = (_Float16*)alloc((size_t)NTOT * 2);
  _Float16* qbuf  = (_Float16*)alloc((size_t)NTOT * 2);
  _Float16* kbuf  = (_Float16*)alloc((size_t)NTOT * 2);
  _Float16* vbuf  = (_Float16*)alloc((size_t)NTOT * 2);
  _Float16* wA    = (_Float16*)alloc((size_t)DD * DD * 2);
  float*    stats = (float*)alloc(128);
  _Float16* headsh = dwh;   // reuse: depthwise buffer dead after conv stack

  dim3 eg(NTOT / 256), eb(256);
  dim3 vg(NTOT / 8 / 256);                    // 8-wide elementwise kernels
  dim3 gg(LL / 64, DD / 64, BB), gb(128);
  dim3 ag(LL / 128, NH, BB), ab(128);
  dim3 cg(DD * DD / 256), cb(256);

  // pos-encoding + LN0
  zero_stats_kernel<<<1, 32, 0, stream>>>(stats);
  add_pos_kernel<<<eg, eb, 0, stream>>>(x, res, stats);
  ln_kernel<<<vg, eb, 0, stream>>>(res, stats, ln_g, ln_b, curh);

  // conv stack
  for (int i = 0; i < 4; ++i) {
    dwconv_kernel<<<vg, eb, 0, stream>>>(curh, dw_w + (size_t)i * DD * KW, dw_b + i * DD, dwh);
    convert_w_kernel<<<cg, cb, 0, stream>>>(pw_w + (size_t)i * DD * DD, wA);
    zero_stats_kernel<<<1, 32, 0, stream>>>(stats);
    gemm_resid_kernel<<<gg, gb, 0, stream>>>(wA, dwh, pw_b + i * DD, res, res, stats, 1);
    ln_kernel<<<vg, eb, 0, stream>>>(res, stats, ln_g + (size_t)(i + 1) * NPB,
                                     ln_b + (size_t)(i + 1) * NPB, curh);
  }

  // QKV projections (1/sqrt(32) folded into Q)
  convert_qkv_w_kernel<<<cg, cb, 0, stream>>>(Wq, wA);
  gemm_half_kernel<<<gg, gb, 0, stream>>>(wA, curh, qbuf, 0.17677669529663687f);
  convert_qkv_w_kernel<<<cg, cb, 0, stream>>>(Wk, wA);
  gemm_half_kernel<<<gg, gb, 0, stream>>>(wA, curh, kbuf, 1.0f);
  convert_qkv_w_kernel<<<cg, cb, 0, stream>>>(Wv, wA);
  gemm_half_kernel<<<gg, gb, 0, stream>>>(wA, curh, vbuf, 1.0f);

  // flash attention
  attn_kernel<<<ag, ab, 0, stream>>>(qbuf, kbuf, vbuf, mask, headsh);

  // Wo projection + residual + LN
  transpose_w_kernel<<<cg, cb, 0, stream>>>(Wo, wA);
  zero_stats_kernel<<<1, 32, 0, stream>>>(stats);
  gemm_resid_kernel<<<gg, gb, 0, stream>>>(wA, headsh, nullptr, res, res, stats, 0);
  ln_kernel<<<vg, eb, 0, stream>>>(res, stats, ln_g + (size_t)5 * NPB,
                                   ln_b + (size_t)5 * NPB, curh);

  // final FC: relu(fc(cur)+b) + res -> d_out (f32)
  convert_w_kernel<<<cg, cb, 0, stream>>>(fc_w, wA);
  gemm_resid_kernel<<<gg, gb, 0, stream>>>(wA, curh, fc_b, res, out, nullptr, 1);
}
